// LinearAutoEncoder_2508260901440
// MI455X (gfx1250) — matrix operand order repaired and checked by reading, not compiled, not run on hardware
//
#include <hip/hip_runtime.h>

// ---------------------------------------------------------------------------
// CDNA5 (gfx1250) 12-layer CUBA-LIF autoencoder.
//   per layer: WMMA GEMM (f16 in / f32 acc) -> LIF scan -> fused axonal delay
// Channel-last intermediate layouts so every WMMA fragment is 2x b128 loads:
//   activations: [B][T][CpadIn] f16     z-buffer: [B][T][CpadOut] f32
// Output channels padded to CpadOut with zero weight rows so every layer
// tiles as 32x32 register blocks: 2 A-frags x 2 B-frags -> 4 WMMAs / 8 loads.
// ---------------------------------------------------------------------------

typedef __attribute__((ext_vector_type(16))) _Float16 v16h;
typedef __attribute__((ext_vector_type(8)))  float    v8f;

#define TT       128            // timesteps
#define NB       16             // batch
#define NTOT     (NB * TT)      // 2048 GEMM columns
#define NGROUPS  (NTOT / 32)    // 64 column-groups of 32

union Frag16 {
    v16h     h;
    uint4    q[2];
    _Float16 f[16];
};
union F32x8 {
    v8f    v;
    float4 q[2];
};

// ---- layer-0 input: [B][C][T] f32 -> [B][T][C] f16 (C = 8192) -------------
__global__ void cvt_x_kernel(const float* __restrict__ x,
                             _Float16* __restrict__ xh, int n) {
    int i = blockIdx.x * blockDim.x + threadIdx.x;
    if (i >= n) return;
    int c  = i & 8191;           // channel (fastest in dst)
    int bt = i >> 13;
    int t  = bt & 127;
    int b  = bt >> 7;
    xh[i] = (_Float16)x[((size_t)b * 8192 + c) * TT + t];
}

// ---- fp32 -> fp16 weight convert, zero-padded in BOTH dims ----------------
// Wh : [CpadOut][CpadIn]
__global__ void cvt_w_kernel(const float* __restrict__ W,
                             _Float16* __restrict__ Wh,
                             int Cout, int Cin, int CpadIn, int CpadOut) {
    int i = blockIdx.x * blockDim.x + threadIdx.x;
    if (i >= CpadOut * CpadIn) return;
    int o = i / CpadIn, k = i % CpadIn;
    Wh[i] = (o < Cout && k < Cin) ? (_Float16)W[(size_t)o * Cin + k]
                                  : (_Float16)0.f;
}

// ---- WMMA GEMM: Z[n, o] = sum_k Wh[o,k] * X[n,k],  n = b*TT + t -----------
// X : [NTOT][CpadIn] f16,  Wh : [CpadOut][CpadIn] f16,  Z : [NTOT][CpadOut]
// One wave: 32(M) x 32(N) register tile, K-chunks of 32.
__global__ void gemm_wmma_kernel(const _Float16* __restrict__ X,
                                 const _Float16* __restrict__ Wh,
                                 float* __restrict__ Z,
                                 int Cpad, int CpO) {
    const int tilesM = CpO >> 5;              // 32-row M tiles
    const int nTiles = tilesM * NGROUPS;
    const int wave   = (int)((blockIdx.x * blockDim.x + threadIdx.x) >> 5);
    if (wave >= nTiles) return;               // wave-uniform; EXEC stays all-1s

    const int tm   = wave % tilesM;
    const int tg   = wave / tilesM;           // column group of 32
    const int lane = threadIdx.x & 31;
    const int g    = lane >> 4;               // K-half select
    const int ln   = lane & 15;

    const int m0 = tm * 32;                   // two 16-row A subtiles
    const int n0 = tg * 32 + ln;              // two 16-col B subtiles
    const int n1 = n0 + 16;

    // A: row; group 0 -> K {0..7,16..23}, group 1 -> {8..15,24..31}
    const _Float16* wr0 = Wh + (size_t)(m0 + ln) * Cpad + g * 8;
    const _Float16* wr1 = wr0 + (size_t)16 * Cpad;
    // B: column n; group 0 -> K {0..7,16..23}, group 1 -> {8..15,24..31}
    const _Float16* xr0 = X + (size_t)n0 * Cpad + g * 8;
    const _Float16* xr1 = X + (size_t)n1 * Cpad + g * 8;

    v8f c00 = {}, c01 = {}, c10 = {}, c11 = {};
    for (int kc = 0; kc < Cpad; kc += 32) {
        __builtin_prefetch(wr0 + kc + 32, 0, 3);   // global_prefetch_b8
        __builtin_prefetch(xr0 + kc + 32, 0, 3);

        Frag16 a0, a1;
        const uint4* pa0 = (const uint4*)(wr0 + kc);
        const uint4* pa1 = (const uint4*)(wr1 + kc);
        a0.q[0] = pa0[0]; a0.q[1] = pa0[2];   // K = kc + g*8 (+16) + {0..7}
        a1.q[0] = pa1[0]; a1.q[1] = pa1[2];

        Frag16 b0, b1;
        const uint4* pb0 = (const uint4*)(xr0 + kc);
        const uint4* pb1 = (const uint4*)(xr1 + kc);
        b0.q[0] = pb0[0]; b0.q[1] = pb0[2];   // K = kc + g*8 (+16) + {0..7}
        b1.q[0] = pb1[0]; b1.q[1] = pb1[2];

        c00 = __builtin_amdgcn_wmma_f32_16x16x32_f16(
                  false, a0.h, false, b0.h, (short)0, c00, false, false);
        c01 = __builtin_amdgcn_wmma_f32_16x16x32_f16(
                  false, a0.h, false, b1.h, (short)0, c01, false, false);
        c10 = __builtin_amdgcn_wmma_f32_16x16x32_f16(
                  false, a1.h, false, b0.h, (short)0, c10, false, false);
        c11 = __builtin_amdgcn_wmma_f32_16x16x32_f16(
                  false, a1.h, false, b1.h, (short)0, c11, false, false);
    }

    // C/D layout: VGPR r -> M = g*8 + r (contiguous channels), N = ln
    F32x8 t00, t01, t10, t11;
    t00.v = c00; t01.v = c01; t10.v = c10; t11.v = c11;
    float* z0 = Z + (size_t)n0 * CpO + m0 + g * 8;
    float* z1 = Z + (size_t)n1 * CpO + m0 + g * 8;
    ((float4*)z0)[0] = t00.q[0];  ((float4*)z0)[1] = t00.q[1];
    ((float4*)(z0 + 16))[0] = t10.q[0];  ((float4*)(z0 + 16))[1] = t10.q[1];
    ((float4*)z1)[0] = t01.q[0];  ((float4*)z1)[1] = t01.q[1];
    ((float4*)(z1 + 16))[0] = t11.q[0];  ((float4*)(z1 + 16))[1] = t11.q[1];
}

// ---- CUBA LIF scan + fused axonal delay -----------------------------------
// Z : [NB][TT][CpO] f32 (channel-last).
// outH : [NB][TT][CpadOut] f16 (next layer input, zero-padded channels)
// outF : [NB][Cout][TT]    f32 (final output, reference layout)
__global__ void lif_kernel(const float* __restrict__ Z,
                           const float* __restrict__ delay,
                           _Float16* __restrict__ outH,
                           float* __restrict__ outF,
                           int Cout, int CpO) {
    const int idx = blockIdx.x * blockDim.x + threadIdx.x;
    if (idx >= NB * CpO) return;
    const int b = idx / CpO;
    const int o = idx % CpO;

    _Float16* rh = outH ? outH + (size_t)b * TT * CpO + o : nullptr;
    if (o >= Cout) {                          // padded channels: all-zero spikes
        if (rh)
            for (int t = 0; t < TT; ++t) rh[(size_t)t * CpO] = (_Float16)0.f;
        return;
    }
    const float* zr = Z + (size_t)b * TT * CpO + o;              // stride CpO
    float* rf = outF ? outF + ((size_t)b * Cout + o) * TT : nullptr;

    int d = (int)rintf(delay[o]);             // round-half-even == jnp.round
    d = d < 0 ? 0 : (d > 62 ? 62 : d);

    for (int t = 0; t < d; ++t) {             // zero prefix from the shift
        if (rh) rh[(size_t)t * CpO] = (_Float16)0.f;
        if (rf) rf[t] = 0.f;
    }

    const float ac = 1.0f - 0.25f;            // current decay
    const float av = 1.0f - 0.03f;            // voltage decay
    float cur = 0.f, volt = 0.f;
    for (int t = 0; t < TT; ++t) {
        cur  = ac * cur + zr[(size_t)t * CpO];
        volt = av * volt + cur;
        float s = (volt >= 1.25f) ? 1.f : 0.f;
        volt = volt * (1.0f - s);
        int td = t + d;
        if (td < TT) {
            if (rh) rh[(size_t)td * CpO] = (_Float16)s;
            if (rf) rf[td] = s;
        }
    }
}

// ---------------------------------------------------------------------------
extern "C" void kernel_launch(void* const* d_in, const int* in_sizes, int n_in,
                              void* d_out, int out_size, void* d_ws, size_t ws_size,
                              hipStream_t stream) {
    (void)in_sizes; (void)n_in; (void)out_size; (void)ws_size;

    static const int dims[13] = {8192,512,256,128,64,32,16,32,64,128,256,512,8192};
    int cpad[13];
    for (int i = 0; i < 13; ++i) cpad[i] = dims[i] < 32 ? 32 : dims[i];

    char* ws = (char*)d_ws;
    const size_t ACT_ELE = (size_t)NB * TT * 8192;            // 16,777,216 elems
    _Float16* act0  = (_Float16*)ws;                          // 32 MB
    _Float16* act1  = (_Float16*)(ws + ACT_ELE * 2);          // 32 MB
    float*    zbuf  = (float*)   (ws + ACT_ELE * 4);          // 64 MB
    _Float16* wbase = (_Float16*)(ws + ACT_ELE * 8);          // ~17.5 MB

    size_t woff[12], acc = 0;
    for (int l = 0; l < 12; ++l) { woff[l] = acc; acc += (size_t)cpad[l+1] * cpad[l]; }

    // layer-0 input: fp32 [B][C][T] -> fp16 [B][T][C]
    {
        int n = (int)ACT_ELE;
        cvt_x_kernel<<<(n + 255) / 256, 256, 0, stream>>>((const float*)d_in[0], act0, n);
    }

    _Float16* acts[2] = {act0, act1};
    for (int l = 0; l < 12; ++l) {
        const int Cin  = dims[l],   Cp  = cpad[l];
        const int Cout = dims[l+1], CpO = cpad[l+1];
        const float* W  = (const float*)d_in[1 + 2 * l];
        const float* dl = (const float*)d_in[2 + 2 * l];
        _Float16* Wh = wbase + woff[l];

        int nw = CpO * Cp;
        cvt_w_kernel<<<(nw + 255) / 256, 256, 0, stream>>>(W, Wh, Cout, Cin, Cp, CpO);

        int tiles  = (CpO >> 5) * NGROUPS;
        int blocks = (tiles + 7) / 8;                          // 8 waves / block
        gemm_wmma_kernel<<<blocks, 256, 0, stream>>>(acts[l & 1], Wh, zbuf, Cp, CpO);

        const bool last = (l == 11);
        int nl = NB * CpO;
        lif_kernel<<<(nl + 255) / 256, 256, 0, stream>>>(
            zbuf, dl,
            last ? nullptr : acts[(l + 1) & 1],
            last ? (float*)d_out : nullptr,
            Cout, CpO);
    }
}
